// GPTBigCodeAttention_17059610100336
// MI455X (gfx1250) — matrix operand order repaired and checked by reading, not compiled, not run on hardware
//
#include <hip/hip_runtime.h>
#include <hip/hip_bf16.h>

// ---------------------------------------------------------------------------
// GPT-BigCode MQA attention for MI455X (gfx1250), bf16 WMMA path.
// B=2, S=2048, D=2048, H=16, HD=128. All matmuls via v_wmma_f32_16x16x32_bf16.
// GEMMs are double-buffered; tile staging uses GLOBAL_LOAD_ASYNC_TO_LDS_B128
// (ASYNCcnt DMA) when the toolchain exposes the builtin, else plain loads.
// ---------------------------------------------------------------------------

typedef __attribute__((ext_vector_type(16))) __bf16 v16bf;
typedef __attribute__((ext_vector_type(8)))  float  v8f;

#if defined(__gfx1250__) &&                                              \
    __has_builtin(__builtin_amdgcn_global_load_async_to_lds_b128) &&     \
    __has_builtin(__builtin_amdgcn_s_wait_asynccnt)
#define HAVE_ASYNC_LDS 1
// Builtin signature (from hipcc diagnostics): param0 = global source,
// 'vector_size(16) int' pointee in addrspace(1); param1 = LDS dest in
// addrspace(3); params 2/3 = imm offset, imm cpol.
typedef int v4i_async __attribute__((vector_size(16)));
#define AS1(p) ((__attribute__((address_space(1))) v4i_async*)(p))
#define AS3(p) ((__attribute__((address_space(3))) v4i_async*)(p))
#else
#define HAVE_ASYNC_LDS 0
#endif

union AFrag {
    v16bf v;
    unsigned int u[8];
    unsigned short s[16];
};

__device__ __forceinline__ unsigned short f32_to_bf16(float f) {
    unsigned int u = __float_as_uint(f);
    unsigned int r = (u + 0x7FFFu + ((u >> 16) & 1u)) >> 16;
    return (unsigned short)r;
}

// A-matrix (16x32 bf16) per-lane K index for VGPR v, lane-half h:
//   v<4 -> 2v, v>=4 -> 2v+8, plus 8*half    (ISA 7.12.2, 16-bit A layout)
__device__ __forceinline__ int kmapA(int v, int half) {
    return ((v < 4) ? (2 * v) : (2 * v + 8)) + 8 * half;
}
// B-matrix (32x16 bf16): same K order as A: v<4 -> 2v, v>=4 -> 2v+8, plus 8*half
__device__ __forceinline__ int kmapB(int v, int half) {
    return ((v < 4) ? (2 * v) : (2 * v + 8)) + 8 * half;
}

// ---------------------------------------------------------------------------
// fp32 -> bf16 conversion (all sizes divisible by 4)
// ---------------------------------------------------------------------------
__global__ void cvt_f32_bf16(const float* __restrict__ in,
                             unsigned short* __restrict__ out, int n) {
    int i = (blockIdx.x * blockDim.x + threadIdx.x) * 4;
    if (i + 3 < n) {
        float4 f = *(const float4*)(in + i);
        ushort4 r;
        r.x = f32_to_bf16(f.x);
        r.y = f32_to_bf16(f.y);
        r.z = f32_to_bf16(f.z);
        r.w = f32_to_bf16(f.w);
        *(ushort4*)(out + i) = r;
    }
}

// ---------------------------------------------------------------------------
// Tiled bf16 GEMM: C[M,N] = A[M,K] @ B[K,N] + bias[N]
// Block tile 128x128, K-step 32 (one WMMA K slab). 256 threads = 8 waves,
// each wave owns a 64x32 sub-tile = 4x2 WMMA accumulators.
// Double-buffered LDS; tile t+1 staged (A via async DMA, B via registers)
// before computing tile t; one barrier per iteration.
// ---------------------------------------------------------------------------
#define G_TM 128
#define G_TN 128
#define G_TK 32
#define G_AS 40   // LDS row stride (bf16) for A tile [128][32]
#define G_BS 40   // LDS row stride (bf16) for Bt tile [128 n][32 k]

template <int OUT_F32>
__global__ __launch_bounds__(256) void gemm_bf16_wmma(
    const unsigned short* __restrict__ A,   // M x K, bf16, row-major
    const unsigned short* __restrict__ Bm,  // K x N, bf16, row-major
    const float* __restrict__ bias,         // N
    void* __restrict__ Cout,                // M x N
    int M, int N, int K) {
    __shared__ unsigned short Ash[2][G_TM * G_AS];
    __shared__ unsigned short Bsh[2][G_TN * G_BS];

    const int tid   = threadIdx.x;
    const int lane  = tid & 31;
    const int wid   = tid >> 5;
    const int ln    = lane & 15;
    const int half  = lane >> 4;
    const int waveM = wid & 1;   // 2 waves over M
    const int waveN = wid >> 1;  // 4 waves over N

    const int mBase = blockIdx.y * G_TM;
    const int nBase = blockIdx.x * G_TN;

    v8f acc[4][2];
    for (int i = 0; i < 4; i++)
        for (int j = 0; j < 2; j++)
            for (int e = 0; e < 8; e++) acc[i][j][e] = 0.f;

    // A-tile staging assignment: thread -> (row, 16-col segment)
    const int arow = tid >> 1;          // 0..127
    const int aseg = (tid & 1) * 16;    // 0 or 16
    // B-tile staging assignment: two 8-elem chunks per thread
    const int bk0r = tid >> 4;          // 0..15
    const int bk1r = (tid + 256) >> 4;  // 16..31
    const int bn0  = (tid & 15) * 8;    // 0..120 (same for both chunks)

    uint4 bstage[2];
#if !HAVE_ASYNC_LDS
    uint4 astage[2];
#endif

    // ---- stage + commit tile 0 ----
    {
        const unsigned short* asrc = A + (size_t)(mBase + arow) * K + aseg;
#if HAVE_ASYNC_LDS
        __builtin_amdgcn_global_load_async_to_lds_b128(
            AS1(asrc), AS3(&Ash[0][arow * G_AS + aseg]), 0, 0);
        __builtin_amdgcn_global_load_async_to_lds_b128(
            AS1(asrc + 8), AS3(&Ash[0][arow * G_AS + aseg + 8]), 0, 0);
#else
        astage[0] = *(const uint4*)asrc;
        astage[1] = *(const uint4*)(asrc + 8);
#endif
        bstage[0] = *(const uint4*)(Bm + (size_t)bk0r * N + nBase + bn0);
        bstage[1] = *(const uint4*)(Bm + (size_t)bk1r * N + nBase + bn0);

#if !HAVE_ASYNC_LDS
        *(uint4*)&Ash[0][arow * G_AS + aseg]     = astage[0];
        *(uint4*)&Ash[0][arow * G_AS + aseg + 8] = astage[1];
#endif
        const unsigned short* e0 = (const unsigned short*)&bstage[0];
        const unsigned short* e1 = (const unsigned short*)&bstage[1];
#pragma unroll
        for (int j = 0; j < 8; j++) {
            Bsh[0][(bn0 + j) * G_BS + bk0r] = e0[j];
            Bsh[0][(bn0 + j) * G_BS + bk1r] = e1[j];
        }
    }
#if HAVE_ASYNC_LDS
    __builtin_amdgcn_s_wait_asynccnt(0);
#endif
    __syncthreads();

    int cur = 0;
    for (int k0 = 0; k0 < K; k0 += G_TK) {
        const int kn  = k0 + G_TK;
        const int nxt = cur ^ 1;

        // --- stage tile t+1 (A: async DMA into LDS, B: regs) ---
        if (kn < K) {
            const unsigned short* asrc = A + (size_t)(mBase + arow) * K + kn + aseg;
#if HAVE_ASYNC_LDS
            __builtin_amdgcn_global_load_async_to_lds_b128(
                AS1(asrc), AS3(&Ash[nxt][arow * G_AS + aseg]), 0, 0);
            __builtin_amdgcn_global_load_async_to_lds_b128(
                AS1(asrc + 8), AS3(&Ash[nxt][arow * G_AS + aseg + 8]), 0, 0);
#else
            astage[0] = *(const uint4*)asrc;
            astage[1] = *(const uint4*)(asrc + 8);
#endif
            bstage[0] = *(const uint4*)(Bm + (size_t)(kn + bk0r) * N + nBase + bn0);
            bstage[1] = *(const uint4*)(Bm + (size_t)(kn + bk1r) * N + nBase + bn0);
        }

        // --- compute tile t from Ash/Bsh[cur] ---
        AFrag af[4], bfr[2];
#pragma unroll
        for (int mt = 0; mt < 4; mt++) {
            int m = waveM * 64 + mt * 16 + ln;
#pragma unroll
            for (int v = 0; v < 8; v++)
                af[mt].u[v] =
                    *(const unsigned int*)&Ash[cur][m * G_AS + kmapA(v, half)];
        }
#pragma unroll
        for (int nt = 0; nt < 2; nt++) {
            int n = waveN * 32 + nt * 16 + ln;
#pragma unroll
            for (int v = 0; v < 8; v++)
                bfr[nt].u[v] =
                    *(const unsigned int*)&Bsh[cur][n * G_BS + kmapB(v, half)];
        }
#pragma unroll
        for (int mt = 0; mt < 4; mt++)
#pragma unroll
            for (int nt = 0; nt < 2; nt++)
                acc[mt][nt] = __builtin_amdgcn_wmma_f32_16x16x32_bf16(
                    false, af[mt].v, false, bfr[nt].v, (short)0, acc[mt][nt],
                    false, false);

        // --- commit tile t+1 staging into the other buffer ---
        if (kn < K) {
#if !HAVE_ASYNC_LDS
            *(uint4*)&Ash[nxt][arow * G_AS + aseg]     = astage[0];
            *(uint4*)&Ash[nxt][arow * G_AS + aseg + 8] = astage[1];
#endif
            const unsigned short* e0 = (const unsigned short*)&bstage[0];
            const unsigned short* e1 = (const unsigned short*)&bstage[1];
#pragma unroll
            for (int j = 0; j < 8; j++) {
                Bsh[nxt][(bn0 + j) * G_BS + bk0r] = e0[j];
                Bsh[nxt][(bn0 + j) * G_BS + bk1r] = e1[j];
            }
        }
#if HAVE_ASYNC_LDS
        __builtin_amdgcn_s_wait_asynccnt(0);
#endif
        __syncthreads();
        cur = nxt;
    }

    // --- epilogue: bias + store ---
#pragma unroll
    for (int mt = 0; mt < 4; mt++) {
#pragma unroll
        for (int nt = 0; nt < 2; nt++) {
            int col = nBase + waveN * 32 + nt * 16 + ln;
            float bv = bias[col];
#pragma unroll
            for (int j = 0; j < 8; j++) {
                int row = mBase + waveM * 64 + mt * 16 + j + 8 * half;
                float vres = acc[mt][nt][j] + bv;
                if (OUT_F32)
                    ((float*)Cout)[(size_t)row * N + col] = vres;
                else
                    ((unsigned short*)Cout)[(size_t)row * N + col] =
                        f32_to_bf16(vres);
            }
        }
    }
}

// ---------------------------------------------------------------------------
// Flash-attention (causal, MQA). qkv: (B*S) x 2304 bf16 rows
//   [0,2048)    Q (16 heads x 128)
//   [2048,2176) K (shared head)
//   [2176,2304) V (shared head)
// Grid: (B*H, S/128). Block: 256 threads; 8 waves x 16 query rows each.
// Q fragments live in VGPRs for the whole key loop. K tile staged via async
// DMA; V tile transposed through registers (overlaps the K DMA).
// ---------------------------------------------------------------------------
#define FA_BQ 128
#define FA_BK 32
#define K_STR 136  // Ksh [32 key][128 hd] row stride
#define V_STR 40   // Vsh [128 hd][32 key] row stride

__global__ __launch_bounds__(256) void flash_attn_mqa(
    const unsigned short* __restrict__ qkv,
    unsigned short* __restrict__ attn_out,  // (B*S) x 2048 bf16
    int Sq, int Hh) {
    __shared__ unsigned short Ksh[FA_BK * K_STR];
    __shared__ unsigned short Vsh[128 * V_STR];
    __shared__ unsigned short Psh[8 * 16 * 32];  // per-wave P scratch

    const int b     = blockIdx.x / Hh;
    const int h     = blockIdx.x % Hh;
    const int qBase = blockIdx.y * FA_BQ;

    const int tid  = threadIdx.x;
    const int lane = tid & 31;
    const int wid  = tid >> 5;
    const int ln   = lane & 15;
    const int half = lane >> 4;

    const int   QKVLD = 2304;
    const float scale = 0.08838834764831845f;  // 1/sqrt(128)

    // --- resident Q fragments for this wave's 16 rows (HD=128 = 4 K-slabs) ---
    const int qRow0 = qBase + wid * 16;
    AFrag aq[4];
    {
        const unsigned short* qptr =
            qkv + (size_t)(b * Sq + qRow0 + ln) * QKVLD + h * 128;
#pragma unroll
        for (int kd = 0; kd < 4; kd++)
#pragma unroll
            for (int v = 0; v < 8; v++)
                aq[kd].u[v] =
                    *(const unsigned int*)(qptr + kd * 32 + kmapA(v, half));
    }

    float mrow[8], lrow[8];
#pragma unroll
    for (int j = 0; j < 8; j++) { mrow[j] = -3.0e38f; lrow[j] = 0.f; }
    v8f oacc[8];
    for (int t = 0; t < 8; t++)
        for (int e = 0; e < 8; e++) oacc[t][e] = 0.f;

    unsigned short* Pw = &Psh[wid * 16 * 32];

    const int kEnd = qBase + FA_BQ;  // causal: skip key blocks past diagonal
    for (int kb = 0; kb < kEnd; kb += FA_BK) {
        // --- K tile (32 keys x 128) -> LDS row-major (async DMA if avail) ---
        {
            int key = tid >> 3;        // 0..31
            int seg = (tid & 7) * 16;  // 0..112
            const unsigned short* src =
                qkv + (size_t)(b * Sq + kb + key) * QKVLD + 2048 + seg;
#if HAVE_ASYNC_LDS
            __builtin_amdgcn_global_load_async_to_lds_b128(
                AS1(src), AS3(&Ksh[key * K_STR + seg]), 0, 0);
            __builtin_amdgcn_global_load_async_to_lds_b128(
                AS1(src + 8), AS3(&Ksh[key * K_STR + seg + 8]), 0, 0);
#else
            uint4 d0 = *(const uint4*)(src);
            uint4 d1 = *(const uint4*)(src + 8);
            *(uint4*)&Ksh[key * K_STR + seg]     = d0;
            *(uint4*)&Ksh[key * K_STR + seg + 8] = d1;
#endif
        }
        // --- V tile (32 keys x 128) -> LDS transposed Vsh[hd][key] ---
        for (int cc = 0; cc < 2; ++cc) {
            int c   = tid + cc * 256;
            int key = c >> 4;
            int n0  = (c & 15) * 8;
            const unsigned short* src =
                qkv + (size_t)(b * Sq + kb + key) * QKVLD + 2176 + n0;
            uint4 d = *(const uint4*)(src);
            const unsigned short* e = (const unsigned short*)&d;
#pragma unroll
            for (int j = 0; j < 8; j++) Vsh[(n0 + j) * V_STR + key] = e[j];
        }
        // --- prefetch next K/V tiles into cache ---
        if (kb + FA_BK < kEnd) {
            const unsigned short* nk =
                qkv + (size_t)(b * Sq + kb + FA_BK + (tid >> 3)) * QKVLD + 2048 +
                (tid & 7) * 16;
            __builtin_prefetch(nk, 0, 1);
            const unsigned short* nv =
                qkv + (size_t)(b * Sq + kb + FA_BK + (tid >> 4)) * QKVLD + 2176 +
                (tid & 15) * 8;
            __builtin_prefetch(nv, 0, 1);
        }
#if HAVE_ASYNC_LDS
        __builtin_amdgcn_s_wait_asynccnt(0);
#endif
        __syncthreads();

        // --- scores S(16q x 32k) = Q @ K^T  (B[k][n] = K[n][k]: contiguous) ---
        v8f sc[2];
        for (int nt = 0; nt < 2; nt++)
            for (int e = 0; e < 8; e++) sc[nt][e] = 0.f;
#pragma unroll
        for (int nt = 0; nt < 2; nt++) {
#pragma unroll
            for (int kd = 0; kd < 4; kd++) {
                AFrag bk;
#pragma unroll
                for (int v = 0; v < 8; v++)
                    bk.u[v] = *(const unsigned int*)&Ksh[(nt * 16 + ln) * K_STR +
                                                         kd * 32 + kmapB(v, half)];
                sc[nt] = __builtin_amdgcn_wmma_f32_16x16x32_bf16(
                    false, aq[kd].v, false, bk.v, (short)0, sc[nt], false, false);
            }
        }

        // --- scale + causal mask + online softmax (per C-fragment row) ---
#pragma unroll
        for (int j = 0; j < 8; j++) {
            int   qr = qRow0 + j + 8 * half;
            float s0 = sc[0][j] * scale;
            float s1 = sc[1][j] * scale;
            if (kb + ln > qr)      s0 = -3.0e38f;
            if (kb + 16 + ln > qr) s1 = -3.0e38f;

            float mx = fmaxf(s0, s1);
#pragma unroll
            for (int d = 1; d < 16; d <<= 1) mx = fmaxf(mx, __shfl_xor(mx, d, 32));
            float mnew = fmaxf(mrow[j], mx);
            float corr = __expf(mrow[j] - mnew);
            float p0   = __expf(s0 - mnew);
            float p1   = __expf(s1 - mnew);
            float ps   = p0 + p1;
#pragma unroll
            for (int d = 1; d < 16; d <<= 1) ps += __shfl_xor(ps, d, 32);
            lrow[j] = lrow[j] * corr + ps;
            mrow[j] = mnew;
#pragma unroll
            for (int t = 0; t < 8; t++) oacc[t][j] *= corr;
            // stash P (bf16) to per-wave LDS for C->A fragment relayout
            Pw[(j + 8 * half) * 32 + ln]      = f32_to_bf16(p0);
            Pw[(j + 8 * half) * 32 + 16 + ln] = f32_to_bf16(p1);
        }

        // --- P A-fragment (same-wave LDS is in-order; no barrier needed) ---
        AFrag ap;
#pragma unroll
        for (int v = 0; v < 8; v++)
            ap.u[v] = *(const unsigned int*)&Pw[ln * 32 + kmapA(v, half)];

        // --- O += P @ V ---
#pragma unroll
        for (int t = 0; t < 8; t++) {
            AFrag bv;
#pragma unroll
            for (int v = 0; v < 8; v++)
                bv.u[v] = *(const unsigned int*)&Vsh[(t * 16 + ln) * V_STR +
                                                     kmapB(v, half)];
            oacc[t] = __builtin_amdgcn_wmma_f32_16x16x32_bf16(
                false, ap.v, false, bv.v, (short)0, oacc[t], false, false);
        }
        __syncthreads();
    }

    // --- normalize + write attn output (bf16, row stride 2048) ---
#pragma unroll
    for (int t = 0; t < 8; t++) {
        int col = h * 128 + t * 16 + ln;
#pragma unroll
        for (int j = 0; j < 8; j++) {
            int   row = qRow0 + j + 8 * half;
            float val = oacc[t][j] / lrow[j];
            attn_out[(size_t)(b * Sq + row) * 2048 + col] = f32_to_bf16(val);
        }
    }
}

// ---------------------------------------------------------------------------
// Launch
// ---------------------------------------------------------------------------
extern "C" void kernel_launch(void* const* d_in, const int* in_sizes, int n_in,
                              void* d_out, int out_size, void* d_ws,
                              size_t ws_size, hipStream_t stream) {
    (void)in_sizes; (void)n_in; (void)out_size; (void)ws_size;
    const float* hs     = (const float*)d_in[0];
    const float* W_attn = (const float*)d_in[1];
    const float* b_attn = (const float*)d_in[2];
    const float* W_proj = (const float*)d_in[3];
    const float* b_proj = (const float*)d_in[4];
    // d_in[5] = attention_mask: causal lower-triangular, handled analytically.

    const int Bb = 2, Ss = 2048, Dd = 2048, Hh = 16, HDh = 128;
    const int BS   = Bb * Ss;            // 4096
    const int NQKV = Hh * HDh + 2 * HDh; // 2304

    char*  ws  = (char*)d_ws;
    size_t off = 0;
    auto alloc = [&](size_t bytes) -> void* {
        void* p = ws + off;
        off += (bytes + 255) & ~(size_t)255;
        return p;
    };
    unsigned short* hsb  = (unsigned short*)alloc((size_t)BS * Dd * 2);    // 16 MB
    unsigned short* Wab  = (unsigned short*)alloc((size_t)Dd * NQKV * 2);  // 9.4 MB
    unsigned short* Wpb  = (unsigned short*)alloc((size_t)Dd * Dd * 2);    // 8 MB
    unsigned short* qkvb = (unsigned short*)alloc((size_t)BS * NQKV * 2);  // 18.9 MB
    unsigned short* attb = (unsigned short*)alloc((size_t)BS * Dd * 2);    // 16 MB

    const int n1 = BS * Dd, n2 = Dd * NQKV, n3 = Dd * Dd;
    cvt_f32_bf16<<<(n1 / 4 + 255) / 256, 256, 0, stream>>>(hs, hsb, n1);
    cvt_f32_bf16<<<(n2 / 4 + 255) / 256, 256, 0, stream>>>(W_attn, Wab, n2);
    cvt_f32_bf16<<<(n3 / 4 + 255) / 256, 256, 0, stream>>>(W_proj, Wpb, n3);

    // QKV projection: (4096 x 2048) @ (2048 x 2304) + b_attn -> bf16
    dim3 g1(NQKV / G_TN, BS / G_TM);  // (18, 32)
    gemm_bf16_wmma<0><<<g1, 256, 0, stream>>>(hsb, Wab, b_attn, qkvb, BS, NQKV, Dd);

    // Causal MQA flash attention
    dim3 gf(Bb * Hh, Ss / FA_BQ);  // (32, 16)
    flash_attn_mqa<<<gf, 256, 0, stream>>>(qkvb, attb, Ss, Hh);

    // Output projection: (4096 x 2048) @ (2048 x 2048) + b_proj -> f32 d_out
    dim3 g2(Dd / G_TN, BS / G_TM);  // (16, 32)
    gemm_bf16_wmma<1><<<g2, 256, 0, stream>>>(attb, Wpb, b_proj, d_out, BS, Dd, Dd);
}
